// GraphTransformer_81286551044270
// MI455X (gfx1250) — matrix operand order repaired and checked by reading, not compiled, not run on hardware
//
#include <hip/hip_runtime.h>

#define NN      50000
#define EE      800000
#define IN_DIM  128
#define HD      256
#define OUT_DIM 128
#define GG      1000

typedef __attribute__((ext_vector_type(16))) __bf16 v16bf;
typedef __attribute__((ext_vector_type(8)))  __bf16 bf16x8;
typedef __attribute__((ext_vector_type(8)))  float  v8f;
typedef __attribute__((ext_vector_type(4)))  unsigned int u32x4;
typedef __attribute__((ext_vector_type(8)))  int i32x8;
typedef __attribute__((ext_vector_type(4)))  int i32x4;

union Frag { v16bf v; bf16x8 h[2]; };

__device__ __forceinline__ int imin(int a, int b) { return a < b ? a : b; }

// ---------------- elementwise helpers ----------------

__global__ void cvt_bf16_kernel(const float* __restrict__ in, __bf16* __restrict__ out, int n) {
  int i = blockIdx.x * blockDim.x + threadIdx.x;
  if (i < n) out[i] = (__bf16)in[i];
}

__global__ void zero_f32_kernel(float* __restrict__ p, int n) {
  int i = blockIdx.x * blockDim.x + threadIdx.x;
  if (i < n) p[i] = 0.0f;
}

// W [K][Nc] f32 (row-major)  ->  Wt [Nc][K] bf16 (row-major)
__global__ void transpose_w_kernel(const float* __restrict__ W, __bf16* __restrict__ Wt,
                                   int K, int Nc) {
  int i = blockIdx.x * blockDim.x + threadIdx.x;
  if (i < K * Nc) {
    int n = i / K, k = i - n * K;
    Wt[i] = (__bf16)W[(size_t)k * Nc + n];
  }
}

// ---------------- WMMA GEMM: C[M][Nc] = A[M][K] * Bt[Nc][K]^T + bias ----------------
// Block = 256 threads = 8 waves; each wave computes a 32x64 tile (8 accumulators).
// The 64xK B tile is staged into LDS once per block by the Tensor Data Mover with
// +16B per-row padding (bank-conflict-free ds_load_b128 fragment reads).

__global__ __launch_bounds__(256) void wmma_gemm_kernel(
    const __bf16* __restrict__ A, const __bf16* __restrict__ Bt,
    const float* __restrict__ bias, float* __restrict__ C,
    __bf16* __restrict__ Cbf, int M, int Nc, int K, int relu)
{
  __shared__ __attribute__((aligned(32))) __bf16 Bs[64 * (256 + 8)];

  const int wave  = threadIdx.x >> 5;
  const int lane  = threadIdx.x & 31;
  const int half  = (lane >> 4) & 1;
  const int r     = lane & 15;
  const int tileM = blockIdx.x * 256 + wave * 32;
  const int tileN = blockIdx.y * 64;
  const int ldb   = K + 8;   // bf16 elements per padded LDS row

  // --- TDM: DMA the 64 x K bf16 tile Bt[tileN .. tileN+63][*] into LDS ---
  if (threadIdx.x < 32) {   // wave 0 issues the tensor DMA
    unsigned lds_base = (unsigned)(unsigned long long)(__bf16*)Bs;
    unsigned long long gaddr =
        (unsigned long long)(const void*)(Bt + (size_t)tileN * K);
    // D# group0: count=1, lds_addr, global_addr, type=2
    u32x4 g0 = { 1u, lds_base, (unsigned)gaddr,
                 (unsigned)(gaddr >> 32) | 0x80000000u };
    // D# group1: data_size=2B, pad_enable, pad every K/2 dwords by 4 dwords (16B),
    //            tensor_dim0=K, tensor_dim1=Nc, tile_dim0=K, tile_dim1=64, dim0_stride=K
    unsigned padi = (K == 256) ? 6u : 5u;            // 128 / 64 dwords per row
    unsigned g1w0 = (1u << 16) | (1u << 20) | (padi << 22) | (3u << 25);
    unsigned g1w1 = ((unsigned)K & 0xFFFFu) << 16;                      // dim0[15:0]
    unsigned g1w2 = (((unsigned)K >> 16) & 0xFFFFu)
                  | (((unsigned)Nc & 0xFFFFu) << 16);                   // dim0[31:16], dim1[15:0]
    unsigned g1w3 = (((unsigned)Nc >> 16) & 0xFFFFu)
                  | (((unsigned)K & 0xFFFFu) << 16);                    // dim1[31:16], tile_dim0
    unsigned g1w4 = 64u;                                                // tile_dim1 (tile_dim2=0)
    unsigned g1w5 = (unsigned)K;                                        // dim0_stride[31:0]
    i32x8 g1 = { (int)g1w0, (int)g1w1, (int)g1w2, (int)g1w3,
                 (int)g1w4, (int)g1w5, 0, 0 };
    i32x4 g2 = { 0, 0, 0, 0 };
    i32x4 g3 = { 0, 0, 0, 0 };
#if defined(__clang_major__) && (__clang_major__ >= 23)
    i32x8 g4 = { 0, 0, 0, 0, 0, 0, 0, 0 };
    __builtin_amdgcn_tensor_load_to_lds(g0, g1, g2, g3, g4, 0);
#else
    __builtin_amdgcn_tensor_load_to_lds(g0, g1, g2, g3, 0);
#endif
    __builtin_amdgcn_s_wait_tensorcnt(0);
  }
  __syncthreads();

  if (tileM < M) {
    v8f acc[2][4] = {};

    int ar0 = imin(tileM + r,      M - 1);   // clamp: keep EXEC full for WMMA
    int ar1 = imin(tileM + 16 + r, M - 1);
    const __bf16* ap0 = A + (size_t)ar0 * K + half * 8;
    const __bf16* ap1 = A + (size_t)ar1 * K + half * 8;

    // prologue: load A fragments for k0 = 0
    Frag afc[2];
    afc[0].h[0] = *(const bf16x8*)(ap0);      afc[0].h[1] = *(const bf16x8*)(ap0 + 16);
    afc[1].h[0] = *(const bf16x8*)(ap1);      afc[1].h[1] = *(const bf16x8*)(ap1 + 16);

    for (int k0 = 0; k0 < K; k0 += 32) {
      // software pipeline: prefetch next K-step's A fragments first
      int kn = (k0 + 32 < K) ? (k0 + 32) : k0;
      Frag afn[2];
      afn[0].h[0] = *(const bf16x8*)(ap0 + kn);      afn[0].h[1] = *(const bf16x8*)(ap0 + kn + 16);
      afn[1].h[0] = *(const bf16x8*)(ap1 + kn);      afn[1].h[1] = *(const bf16x8*)(ap1 + kn + 16);

#pragma unroll
      for (int g = 0; g < 4; g++) {
        // B fragment from LDS: lanes 0-15 K=k0..k0+7,k0+16..k0+23; lanes 16-31 K=k0+8..k0+15,k0+24..k0+31
        Frag bfr;
        const __bf16* bp = &Bs[(g * 16 + r) * ldb + k0 + half * 8];
        bfr.h[0] = *(const bf16x8*)(bp);
        bfr.h[1] = *(const bf16x8*)(bp + 16);
        acc[0][g] = __builtin_amdgcn_wmma_f32_16x16x32_bf16(
            false, afc[0].v, false, bfr.v, (short)0, acc[0][g], false, false);
        acc[1][g] = __builtin_amdgcn_wmma_f32_16x16x32_bf16(
            false, afc[1].v, false, bfr.v, (short)0, acc[1][g], false, false);
      }
      afc[0] = afn[0];
      afc[1] = afn[1];
    }

#pragma unroll
    for (int m = 0; m < 2; m++) {
#pragma unroll
      for (int g = 0; g < 4; g++) {
        int col = tileN + g * 16 + r;
        float bv = bias[col];
#pragma unroll
        for (int i = 0; i < 8; i++) {
          int row = tileM + m * 16 + half * 8 + i;
          if (row < M) {
            float val = acc[m][g][i] + bv;
            if (relu) val = fmaxf(val, 0.0f);
            C[(size_t)row * Nc + col] = val;
            if (Cbf) Cbf[(size_t)row * Nc + col] = (__bf16)val;
          }
        }
      }
    }
  }
}

// ---------------- edge phase ----------------

__device__ __forceinline__ unsigned fmap_ord(float f) {
  unsigned u = __float_as_uint(f);
  return (u & 0x80000000u) ? ~u : (u | 0x80000000u);
}
__device__ __forceinline__ float funmap_ord(unsigned u) {
  return __uint_as_float((u & 0x80000000u) ? (u & 0x7FFFFFFFu) : ~u);
}

// one wave per edge: score[e] = dot(q[dst], k[src]) / 16
__global__ void edge_score_kernel(const float* __restrict__ q, const float* __restrict__ k,
                                  const int* __restrict__ src, const int* __restrict__ dst,
                                  float* __restrict__ score, int nE) {
  int wid  = (blockIdx.x * blockDim.x + threadIdx.x) >> 5;
  int lane = threadIdx.x & 31;
  if (wid >= nE) return;
  int s = src[wid], d = dst[wid];
  const float4* qp = (const float4*)(q + (size_t)d * HD) + lane * 2;
  const float4* kp = (const float4*)(k + (size_t)s * HD) + lane * 2;
  float4 q0 = qp[0], q1 = qp[1];
  float4 k0 = kp[0], k1 = kp[1];
  float acc = q0.x * k0.x + q0.y * k0.y + q0.z * k0.z + q0.w * k0.w
            + q1.x * k1.x + q1.y * k1.y + q1.z * k1.z + q1.w * k1.w;
#pragma unroll
  for (int off = 16; off > 0; off >>= 1) acc += __shfl_xor(acc, off, 32);
  if (lane == 0) score[wid] = acc * 0.0625f;   // 1/sqrt(256)
}

__global__ void init_ms_kernel(unsigned* __restrict__ m, float* __restrict__ s, int n) {
  int i = blockIdx.x * blockDim.x + threadIdx.x;
  if (i < n) { m[i] = 0x007FFFFFu; /* fmap_ord(-inf) */ s[i] = 0.0f; }
}

__global__ void edge_max_kernel(const float* __restrict__ score, const int* __restrict__ dst,
                                unsigned* __restrict__ m, int nE) {
  int e = blockIdx.x * blockDim.x + threadIdx.x;
  if (e < nE) atomicMax(&m[dst[e]], fmap_ord(score[e]));
}

__global__ void edge_expsum_kernel(float* __restrict__ score, const int* __restrict__ dst,
                                   const unsigned* __restrict__ m, float* __restrict__ s, int nE) {
  int e = blockIdx.x * blockDim.x + threadIdx.x;
  if (e < nE) {
    int d = dst[e];
    float ev = __expf(score[e] - funmap_ord(m[d]));
    score[e] = ev;
    unsafeAtomicAdd(&s[d], ev);
  }
}

// one wave per edge: agg[dst] += v[src] * alpha   (agg pre-seeded with h@Ws+bs)
__global__ void edge_agg_kernel(const float* __restrict__ eval, const float* __restrict__ s,
                                const float* __restrict__ v, const int* __restrict__ src,
                                const int* __restrict__ dst, float* __restrict__ agg, int nE) {
  int wid  = (blockIdx.x * blockDim.x + threadIdx.x) >> 5;
  int lane = threadIdx.x & 31;
  if (wid >= nE) return;
  int sn = src[wid], d = dst[wid];
  float alpha = eval[wid] / (s[d] + 1e-16f);
  const float4* vp = (const float4*)(v + (size_t)sn * HD) + lane * 2;
  float4 v0 = vp[0], v1 = vp[1];
  float* ap = agg + (size_t)d * HD + lane * 8;
  unsafeAtomicAdd(ap + 0, v0.x * alpha);
  unsafeAtomicAdd(ap + 1, v0.y * alpha);
  unsafeAtomicAdd(ap + 2, v0.z * alpha);
  unsafeAtomicAdd(ap + 3, v0.w * alpha);
  unsafeAtomicAdd(ap + 4, v1.x * alpha);
  unsafeAtomicAdd(ap + 5, v1.y * alpha);
  unsafeAtomicAdd(ap + 6, v1.z * alpha);
  unsafeAtomicAdd(ap + 7, v1.w * alpha);
}

__global__ void combine_relu_kernel(const float* __restrict__ agg, float* __restrict__ h,
                                    __bf16* __restrict__ hbf, int n) {
  int i = blockIdx.x * blockDim.x + threadIdx.x;
  if (i < n) {
    float val = fmaxf(agg[i], 0.0f);
    h[i] = val;
    hbf[i] = (__bf16)val;
  }
}

// ---------------- pooling + epilogue ----------------

__global__ void pool_count_kernel(const int* __restrict__ batch, float* __restrict__ counts, int n) {
  int i = blockIdx.x * blockDim.x + threadIdx.x;
  if (i < n) unsafeAtomicAdd(&counts[batch[i]], 1.0f);
}

// one wave per node: pooled[batch[n]] += h[n]
__global__ void pool_sum_kernel(const float* __restrict__ h, const int* __restrict__ batch,
                                float* __restrict__ pooled, int n) {
  int wid  = (blockIdx.x * blockDim.x + threadIdx.x) >> 5;
  int lane = threadIdx.x & 31;
  if (wid >= n) return;
  int g = batch[wid];
  const float4* hp = (const float4*)(h + (size_t)wid * HD) + lane * 2;
  float4 h0 = hp[0], h1 = hp[1];
  float* pp = pooled + (size_t)g * HD + lane * 8;
  unsafeAtomicAdd(pp + 0, h0.x); unsafeAtomicAdd(pp + 1, h0.y);
  unsafeAtomicAdd(pp + 2, h0.z); unsafeAtomicAdd(pp + 3, h0.w);
  unsafeAtomicAdd(pp + 4, h1.x); unsafeAtomicAdd(pp + 5, h1.y);
  unsafeAtomicAdd(pp + 6, h1.z); unsafeAtomicAdd(pp + 7, h1.w);
}

__global__ void pool_div_cvt_kernel(float* __restrict__ pooled, const float* __restrict__ counts,
                                    __bf16* __restrict__ out, int n, int Hdim) {
  int i = blockIdx.x * blockDim.x + threadIdx.x;
  if (i < n) {
    int g = i / Hdim;
    float val = pooled[i] / fmaxf(counts[g], 1.0f);
    pooled[i] = val;
    out[i] = (__bf16)val;
  }
}

// ---------------- driver ----------------

extern "C" void kernel_launch(void* const* d_in, const int* in_sizes, int n_in,
                              void* d_out, int out_size, void* d_ws, size_t ws_size,
                              hipStream_t stream) {
  (void)in_sizes; (void)n_in; (void)out_size; (void)ws_size;

  const float* x     = (const float*)d_in[0];
  const int*   edge  = (const int*)d_in[1];
  const int*   batch = (const int*)d_in[2];
  const int*   srcI  = edge;        // edge_index[0]
  const int*   dstI  = edge + EE;   // edge_index[1]

  const float* cw[3][8];
  for (int l = 0; l < 3; l++)
    for (int j = 0; j < 8; j++)
      cw[l][j] = (const float*)d_in[3 + l * 8 + j];
  const float* W1 = (const float*)d_in[27];
  const float* b1 = (const float*)d_in[28];
  const float* W2 = (const float*)d_in[29];
  const float* b2 = (const float*)d_in[30];

  char* base = (char*)d_ws;
  size_t off = 0;
  auto alloc = [&](size_t bytes) -> void* {
    void* p = base + off;
    off = (off + bytes + 255) & ~(size_t)255;
    return p;
  };
  __bf16*   hbf    = (__bf16*)  alloc((size_t)NN * HD * 2);
  float*    hbuff  = (float*)   alloc((size_t)NN * HD * 4);
  float*    qb     = (float*)   alloc((size_t)NN * HD * 4);
  float*    kb     = (float*)   alloc((size_t)NN * HD * 4);
  float*    vb     = (float*)   alloc((size_t)NN * HD * 4);
  float*    sc     = (float*)   alloc((size_t)EE * 4);
  unsigned* mb     = (unsigned*)alloc((size_t)NN * 4);
  float*    sb     = (float*)   alloc((size_t)NN * 4);
  __bf16*   wt     = (__bf16*)  alloc((size_t)HD * HD * 2);
  float*    pooled = (float*)   alloc((size_t)GG * HD * 4);
  float*    counts = (float*)   alloc((size_t)GG * 4);
  __bf16*   poolbf = (__bf16*)  alloc((size_t)GG * HD * 2);
  float*    mlph   = (float*)   alloc((size_t)GG * HD * 4);
  __bf16*   mlphbf = (__bf16*)  alloc((size_t)GG * HD * 2);
  float*    aggb   = kb;   // k is dead after edge_score; reuse as agg (stream-ordered)

  const int T = 256;

  // x (f32) -> hbf (bf16)
  cvt_bf16_kernel<<<(NN * IN_DIM + T - 1) / T, T, 0, stream>>>(x, hbf, NN * IN_DIM);

  int ci = IN_DIM;
  for (int l = 0; l < 3; l++) {
    const float *Wq = cw[l][0], *bq = cw[l][1], *Wk = cw[l][2], *bk = cw[l][3],
                *Wv = cw[l][4], *bv = cw[l][5], *Ws = cw[l][6], *bs = cw[l][7];
    dim3 gg((NN + 255) / 256, HD / 64);
    int wtN = ci * HD;

    transpose_w_kernel<<<(wtN + T - 1) / T, T, 0, stream>>>(Wq, wt, ci, HD);
    wmma_gemm_kernel<<<gg, T, 0, stream>>>(hbf, wt, bq, qb, nullptr, NN, HD, ci, 0);
    transpose_w_kernel<<<(wtN + T - 1) / T, T, 0, stream>>>(Wk, wt, ci, HD);
    wmma_gemm_kernel<<<gg, T, 0, stream>>>(hbf, wt, bk, kb, nullptr, NN, HD, ci, 0);
    transpose_w_kernel<<<(wtN + T - 1) / T, T, 0, stream>>>(Wv, wt, ci, HD);
    wmma_gemm_kernel<<<gg, T, 0, stream>>>(hbf, wt, bv, vb, nullptr, NN, HD, ci, 0);

    edge_score_kernel<<<(EE * 32 + T - 1) / T, T, 0, stream>>>(qb, kb, srcI, dstI, sc, EE);

    // seed agg with root skip h@Ws + bs (overwrites kb after edge_score consumed it)
    transpose_w_kernel<<<(wtN + T - 1) / T, T, 0, stream>>>(Ws, wt, ci, HD);
    wmma_gemm_kernel<<<gg, T, 0, stream>>>(hbf, wt, bs, aggb, nullptr, NN, HD, ci, 0);

    init_ms_kernel<<<(NN + T - 1) / T, T, 0, stream>>>(mb, sb, NN);
    edge_max_kernel<<<(EE + T - 1) / T, T, 0, stream>>>(sc, dstI, mb, EE);
    edge_expsum_kernel<<<(EE + T - 1) / T, T, 0, stream>>>(sc, dstI, mb, sb, EE);
    edge_agg_kernel<<<(EE * 32 + T - 1) / T, T, 0, stream>>>(sc, sb, vb, srcI, dstI, aggb, EE);
    combine_relu_kernel<<<(NN * HD + T - 1) / T, T, 0, stream>>>(aggb, hbuff, hbf, NN * HD);
    ci = HD;
  }

  // global mean pool
  zero_f32_kernel<<<(GG * HD + T - 1) / T, T, 0, stream>>>(pooled, GG * HD);
  zero_f32_kernel<<<(GG + T - 1) / T, T, 0, stream>>>(counts, GG);
  pool_count_kernel<<<(NN + T - 1) / T, T, 0, stream>>>(batch, counts, NN);
  pool_sum_kernel<<<(NN * 32 + T - 1) / T, T, 0, stream>>>(hbuff, batch, pooled, NN);
  pool_div_cvt_kernel<<<(GG * HD + T - 1) / T, T, 0, stream>>>(pooled, counts, poolbf, GG * HD, HD);

  // MLP head
  dim3 g1((GG + 255) / 256, HD / 64);
  transpose_w_kernel<<<(HD * HD + T - 1) / T, T, 0, stream>>>(W1, wt, HD, HD);
  wmma_gemm_kernel<<<g1, T, 0, stream>>>(poolbf, wt, b1, mlph, mlphbf, GG, HD, HD, 1);

  dim3 g2((GG + 255) / 256, OUT_DIM / 64);
  transpose_w_kernel<<<(HD * OUT_DIM + T - 1) / T, T, 0, stream>>>(W2, wt, HD, OUT_DIM);
  wmma_gemm_kernel<<<g2, T, 0, stream>>>(mlphbf, wt, b2, (float*)d_out, nullptr, GG, OUT_DIM, HD, 0);
}
